// MultiHeadAttention_6219112645140
// MI455X (gfx1250) — matrix operand order repaired and checked by reading, not compiled, not run on hardware
//
#include <hip/hip_runtime.h>

// ---------------------------------------------------------------- constants
constexpr int kB  = 4;
constexpr int kT  = 2048;
constexpr int kC  = 1024;
constexpr int kNH = 16;
constexpr int kHS = 64;   // kC / kNH

typedef __attribute__((ext_vector_type(8)))  _Float16 v8h;
typedef __attribute__((ext_vector_type(16))) _Float16 v16h;
typedef __attribute__((ext_vector_type(8)))  float    v8f;
typedef __attribute__((ext_vector_type(4)))  unsigned int u32x4;
typedef __attribute__((ext_vector_type(8)))  unsigned int u32x8;

#define USE_TDM 1

union HFrag { v16h v; v8h h[2]; };

// Load a 16-f16 WMMA fragment as two contiguous 16-byte LDS pieces.
__device__ inline v16h load_frag2(const _Float16* p0, const _Float16* p1) {
  HFrag f;
  f.h[0] = *(const v8h*)p0;
  f.h[1] = *(const v8h*)p1;
  return f.v;
}

__device__ inline v8f wmma_f16(v16h a, v16h b, v8f c) {
  // D = A(16x32 f16) x B(32x16 f16) + C(16x16 f32)
  return __builtin_amdgcn_wmma_f32_16x16x32_f16(
      false, a, false, b, (short)0, c, false, false);
}

__device__ inline void wait_tensorcnt0() {
#if __has_builtin(__builtin_amdgcn_s_wait_tensorcnt)
  __builtin_amdgcn_s_wait_tensorcnt(0);
#else
  asm volatile("s_wait_tensorcnt 0x0" ::: "memory");
#endif
}

// ---------------------------------------------------------------------------
// Tensor Data Mover: 2D tile load (f16 elements) global -> LDS.
// Builds a D# per CDNA5 ISA and issues
// tensor_load_to_lds with descriptor groups 0 and 1 (2D tensor => VADDR2/3
// omitted / NULL).  Issue from ONE wave; EXEC is ignored by TDM.
// ---------------------------------------------------------------------------
__device__ inline void tdm_load_2d_f16(const void* gptr, void* lds_ptr,
                                       unsigned rows, unsigned cols,
                                       unsigned row_stride) {
#if USE_TDM
  union { u32x4 v; unsigned long long q[2]; } g0;
  union { u32x8 v; unsigned long long q[4]; } g1;
  unsigned long long ga  = (unsigned long long)(size_t)gptr;
  unsigned long long lds = (unsigned int)(size_t)lds_ptr;   // low 32 = LDS byte addr
  // group0: [1:0]=count(1), [63:32]=lds_addr, [120:64]=global_addr, [127:126]=type(2)
  g0.q[0] = 1ull | (lds << 32);
  g0.q[1] = (ga & 0x01FFFFFFFFFFFFFFull) | (2ull << 62);
  // group1: [17:16]=data_size(1=2B), [79:48]=tensor_dim0, [111:80]=tensor_dim1,
  //         [127:112]=tile_dim0, [143:128]=tile_dim1, [207:160]=tensor_dim0_stride
  g1.q[0] = (1ull << 16) | ((unsigned long long)cols << 48);
  g1.q[1] = ((unsigned long long)rows << 16) | ((unsigned long long)cols << 48);
  g1.q[2] = (unsigned long long)rows | ((unsigned long long)row_stride << 32);
  g1.q[3] = 0ull;
  asm volatile("tensor_load_to_lds %0, %1" : : "s"(g0.v), "s"(g1.v) : "memory");
#else
  (void)gptr; (void)lds_ptr; (void)rows; (void)cols; (void)row_stride;
#endif
}

// =====================================================================
// Kernel 1: Y = X(f32)[8192,1024] @ W(f32)[1024,1024] + bias, f16 output
// in head-split layout [B, NH, T, HS].
// 256 threads (8 waves); tile 128x64; K-step 64 (8 WMMAs / barrier).
// =====================================================================
__global__ __launch_bounds__(256)
void qkv_proj_kernel(const float* __restrict__ X, const float* __restrict__ W,
                     const float* __restrict__ bias, _Float16* __restrict__ Yh) {
  __shared__ _Float16 As[128][64];
  __shared__ _Float16 Wt[64][64];   // W tile transposed: Wt[n][k]

  const int tid  = threadIdx.x;
  const int lane = tid & 31;
  const int wid  = tid >> 5;
  const int lh   = lane >> 4;
  const int l16  = lane & 15;
  const int n0   = blockIdx.x * 64;
  const int m0   = blockIdx.y * 128;

  v8f acc[4] = {};

  for (int k0 = 0; k0 < kC; k0 += 64) {
    // ---- X tile (f32 -> f16): 128 x 64, 32 floats / thread
    {
      int row = tid >> 1;
      int cb  = (tid & 1) * 32;
      const float* src = X + (size_t)(m0 + row) * kC + k0 + cb;
      if (k0 + 64 < kC) __builtin_prefetch(src + 64, 0, 0);
      _Float16* dst = &As[row][cb];
#pragma unroll
      for (int j = 0; j < 32; j += 4) {
        float4 f = *(const float4*)(src + j);
        dst[j + 0] = (_Float16)f.x;
        dst[j + 1] = (_Float16)f.y;
        dst[j + 2] = (_Float16)f.z;
        dst[j + 3] = (_Float16)f.w;
      }
    }
    // ---- W tile (f32 -> f16, transposed): 64 x 64, 16 floats / thread
    {
      int kk = tid & 63;
      int nb = (tid >> 6) * 16;
      const float* src = W + (size_t)(k0 + kk) * kC + n0 + nb;
      if (k0 + 64 < kC) __builtin_prefetch(src + (size_t)64 * kC, 0, 0);
#pragma unroll
      for (int j = 0; j < 16; ++j) Wt[nb + j][kk] = (_Float16)src[j];
    }
    __syncthreads();

    const int arow = wid * 16 + l16;
    v16h a[2], b[4][2];
#pragma unroll
    for (int kc = 0; kc < 2; ++kc)
      a[kc] = load_frag2(&As[arow][kc * 32 + lh * 8],
                         &As[arow][kc * 32 + 16 + lh * 8]);
#pragma unroll
    for (int nt = 0; nt < 4; ++nt) {
      int n = nt * 16 + l16;
#pragma unroll
      for (int kc = 0; kc < 2; ++kc)
        b[nt][kc] = load_frag2(&Wt[n][kc * 32 + lh * 8],
                               &Wt[n][kc * 32 + 16 + lh * 8]);
    }
#pragma unroll
    for (int nt = 0; nt < 4; ++nt) {
      acc[nt] = wmma_f16(a[0], b[nt][0], acc[nt]);
      acc[nt] = wmma_f16(a[1], b[nt][1], acc[nt]);
    }
    __syncthreads();
  }

  // ---- epilogue: bias add, f16 store in [B, NH, T, HS]
#pragma unroll
  for (int nt = 0; nt < 4; ++nt) {
    int col = n0 + nt * 16 + l16;
    float bv = bias[col];
    int h = col >> 6;
    int d = col & 63;
#pragma unroll
    for (int r = 0; r < 8; ++r) {
      int grow = m0 + wid * 16 + r + 8 * lh;
      int bb = grow >> 11;
      int t  = grow & 2047;
      Yh[((size_t)(bb * kNH + h) * kT + t) * kHS + d] =
          (_Float16)(acc[nt][r] + bv);
    }
  }
}

// =====================================================================
// Kernel 2: causal flash attention, f16 q/k/v in [B, NH, T, HS].
// 128 threads = 4 waves; each wave owns 16 query rows (block: 64).
// Streams 64-key chunks (16 WMMAs / chunk); K-chunk and Q-tile fetched
// by the Tensor Data Mover; online softmax in f32.
// =====================================================================
__global__ __launch_bounds__(128)
void attn_kernel(const _Float16* __restrict__ Q, const _Float16* __restrict__ K,
                 const _Float16* __restrict__ V, _Float16* __restrict__ O) {
  __shared__ _Float16 Ks[64][64];       // K chunk, row-major [key][d]  (TDM)
  __shared__ _Float16 Vt[64][64];       // V chunk transposed [d][key]
  __shared__ _Float16 Qs[64][64];       // query tile (TDM)
  __shared__ _Float16 Ps[4][16][64];    // per-wave P staging (D->A relayout)

  const int tid  = threadIdx.x;
  const int lane = tid & 31;
  const int wid  = tid >> 5;
  const int lh   = lane >> 4;
  const int l16  = lane & 15;
  const int bh   = blockIdx.y;              // b*NH + h
  const int q0   = blockIdx.x * 64;

  const _Float16* Qg = Q + (size_t)bh * kT * kHS;
  const _Float16* Kg = K + (size_t)bh * kT * kHS;
  const _Float16* Vg = V + (size_t)bh * kT * kHS;

  // ---- Q tile 64x64 via TDM (one wave issues; EXEC ignored by TDM)
  if (wid == 0) {
    tdm_load_2d_f16(Qg + (size_t)q0 * kHS, &Qs[0][0], 64, 64, 64);
    wait_tensorcnt0();
  }
  __syncthreads();

  v16h aq[2];
#pragma unroll
  for (int dc = 0; dc < 2; ++dc)
    aq[dc] = load_frag2(&Qs[wid * 16 + l16][dc * 32 + lh * 8],
                        &Qs[wid * 16 + l16][dc * 32 + 16 + lh * 8]);

  float m[8], l[8];
#pragma unroll
  for (int r = 0; r < 8; ++r) { m[r] = -1e30f; l[r] = 0.0f; }
  v8f o[4] = {};

  const int nchunks  = blockIdx.x + 1;      // causal bound (64-key chunks)
  const float qk_scale = 0.125f;            // 1/sqrt(HS)

  for (int c = 0; c < nchunks; ++c) {
    const int kb = c * 64;
    // ---- K chunk 64x64 via TDM
    if (wid == 0)
      tdm_load_2d_f16(Kg + (size_t)kb * kHS, &Ks[0][0], 64, 64, 64);
    // ---- V chunk load, transposed into Vt[d][key]; 32 f16 / thread
    {
      int kk = tid & 63;
      int db = (tid >> 6) * 32;
      const _Float16* src = Vg + (size_t)(kb + kk) * kHS + db;
      v8h x0 = *(const v8h*)(src);
      v8h x1 = *(const v8h*)(src + 8);
      v8h x2 = *(const v8h*)(src + 16);
      v8h x3 = *(const v8h*)(src + 24);
#pragma unroll
      for (int j = 0; j < 8; ++j) Vt[db + j][kk]      = x0[j];
#pragma unroll
      for (int j = 0; j < 8; ++j) Vt[db + 8 + j][kk]  = x1[j];
#pragma unroll
      for (int j = 0; j < 8; ++j) Vt[db + 16 + j][kk] = x2[j];
#pragma unroll
      for (int j = 0; j < 8; ++j) Vt[db + 24 + j][kk] = x3[j];
      // prefetch next chunk into GL2
      if (c + 1 < nchunks) {
        __builtin_prefetch(src + (size_t)64 * kHS, 0, 0);
        __builtin_prefetch(Kg + (size_t)(kb + 64 + kk) * kHS, 0, 0);
      }
    }
    if (wid == 0) wait_tensorcnt0();
    __syncthreads();

    // ---- S = Q K^T : four 16-key subtiles (B operand = K rows, N-major)
    v8f s[4] = {};
#pragma unroll
    for (int ns = 0; ns < 4; ++ns) {
      int n = ns * 16 + l16;
#pragma unroll
      for (int dc = 0; dc < 2; ++dc) {
        v16h bk = load_frag2(&Ks[n][dc * 32 + lh * 8],
                             &Ks[n][dc * 32 + 16 + lh * 8]);
        s[ns] = wmma_f16(aq[dc], bk, s[ns]);
      }
    }

    // ---- scale + causal mask + chunk row-max (reduce over 16 lanes)
    float rowmax[8];
#pragma unroll
    for (int r = 0; r < 8; ++r) {
      int qi = q0 + wid * 16 + r + 8 * lh;
      float mx = -1e30f;
#pragma unroll
      for (int ns = 0; ns < 4; ++ns) {
        int ki = kb + ns * 16 + l16;
        float sv = s[ns][r] * qk_scale + ((ki > qi) ? -1e9f : 0.0f);
        s[ns][r] = sv;
        mx = fmaxf(mx, sv);
      }
      mx = fmaxf(mx, __shfl_xor(mx, 1, 32));
      mx = fmaxf(mx, __shfl_xor(mx, 2, 32));
      mx = fmaxf(mx, __shfl_xor(mx, 4, 32));
      mx = fmaxf(mx, __shfl_xor(mx, 8, 32));
      rowmax[r] = mx;
    }

    // ---- online softmax update + stage P (f16) into LDS
#pragma unroll
    for (int r = 0; r < 8; ++r) {
      float mn   = fmaxf(m[r], rowmax[r]);
      float corr = __expf(m[r] - mn);
      float p[4], rs = 0.0f;
#pragma unroll
      for (int ns = 0; ns < 4; ++ns) { p[ns] = __expf(s[ns][r] - mn); rs += p[ns]; }
      rs += __shfl_xor(rs, 1, 32);
      rs += __shfl_xor(rs, 2, 32);
      rs += __shfl_xor(rs, 4, 32);
      rs += __shfl_xor(rs, 8, 32);
      l[r] = l[r] * corr + rs;
      m[r] = mn;
#pragma unroll
      for (int nd = 0; nd < 4; ++nd) o[nd][r] *= corr;
      int prow = r + 8 * lh;
#pragma unroll
      for (int ns = 0; ns < 4; ++ns)
        Ps[wid][prow][ns * 16 + l16] = (_Float16)p[ns];
    }

    // ---- O += P @ V  (two K=32 halves over the 64-key chunk)
    v16h ap0 = load_frag2(&Ps[wid][l16][lh * 8],      &Ps[wid][l16][16 + lh * 8]);
    v16h ap1 = load_frag2(&Ps[wid][l16][32 + lh * 8], &Ps[wid][l16][48 + lh * 8]);
#pragma unroll
    for (int nd = 0; nd < 4; ++nd) {
      int dcol = nd * 16 + l16;
      v16h bv0 = load_frag2(&Vt[dcol][lh * 8],      &Vt[dcol][16 + lh * 8]);
      v16h bv1 = load_frag2(&Vt[dcol][32 + lh * 8], &Vt[dcol][48 + lh * 8]);
      o[nd] = wmma_f16(ap0, bv0, o[nd]);
      o[nd] = wmma_f16(ap1, bv1, o[nd]);
    }
    __syncthreads();   // protect Ks/Vt before next chunk overwrite
  }

  // ---- epilogue: divide by softmax denom, f16 store into [B, T, C]
  const int b = bh >> 4;
  const int h = bh & 15;
#pragma unroll
  for (int nd = 0; nd < 4; ++nd) {
    int d = nd * 16 + l16;
#pragma unroll
    for (int r = 0; r < 8; ++r) {
      int qi = q0 + wid * 16 + r + 8 * lh;
      float val = o[nd][r] / l[r];
      O[(size_t)(b * kT + qi) * kC + h * kHS + d] = (_Float16)val;
    }
  }
}

// =====================================================================
// Kernel 3: Out(f32) = A(f16)[8192,1024] @ Wo(f32->f16) + bo.
// Same tiling as kernel 1; f32 output, flat [B*T, C].
// =====================================================================
__global__ __launch_bounds__(256)
void out_proj_kernel(const _Float16* __restrict__ X, const float* __restrict__ W,
                     const float* __restrict__ bias, float* __restrict__ Out) {
  __shared__ _Float16 As[128][64];
  __shared__ _Float16 Wt[64][64];

  const int tid  = threadIdx.x;
  const int lane = tid & 31;
  const int wid  = tid >> 5;
  const int lh   = lane >> 4;
  const int l16  = lane & 15;
  const int n0   = blockIdx.x * 64;
  const int m0   = blockIdx.y * 128;

  v8f acc[4] = {};

  for (int k0 = 0; k0 < kC; k0 += 64) {
    {
      int row = tid >> 1;
      int cb  = (tid & 1) * 32;
      const _Float16* src = X + (size_t)(m0 + row) * kC + k0 + cb;
      if (k0 + 64 < kC) __builtin_prefetch(src + 64, 0, 0);
      *(v8h*)&As[row][cb]      = *(const v8h*)(src);
      *(v8h*)&As[row][cb + 8]  = *(const v8h*)(src + 8);
      *(v8h*)&As[row][cb + 16] = *(const v8h*)(src + 16);
      *(v8h*)&As[row][cb + 24] = *(const v8h*)(src + 24);
    }
    {
      int kk = tid & 63;
      int nb = (tid >> 6) * 16;
      const float* src = W + (size_t)(k0 + kk) * kC + n0 + nb;
      if (k0 + 64 < kC) __builtin_prefetch(src + (size_t)64 * kC, 0, 0);
#pragma unroll
      for (int j = 0; j < 16; ++j) Wt[nb + j][kk] = (_Float16)src[j];
    }
    __syncthreads();

    const int arow = wid * 16 + l16;
    v16h a[2], b[4][2];
#pragma unroll
    for (int kc = 0; kc < 2; ++kc)
      a[kc] = load_frag2(&As[arow][kc * 32 + lh * 8],
                         &As[arow][kc * 32 + 16 + lh * 8]);
#pragma unroll
    for (int nt = 0; nt < 4; ++nt) {
      int n = nt * 16 + l16;
#pragma unroll
      for (int kc = 0; kc < 2; ++kc)
        b[nt][kc] = load_frag2(&Wt[n][kc * 32 + lh * 8],
                               &Wt[n][kc * 32 + 16 + lh * 8]);
    }
#pragma unroll
    for (int nt = 0; nt < 4; ++nt) {
      acc[nt] = wmma_f16(a[0], b[nt][0], acc[nt]);
      acc[nt] = wmma_f16(a[1], b[nt][1], acc[nt]);
    }
    __syncthreads();
  }

#pragma unroll
  for (int nt = 0; nt < 4; ++nt) {
    int col = n0 + nt * 16 + l16;
    float bv = bias[col];
#pragma unroll
    for (int r = 0; r < 8; ++r) {
      int grow = m0 + wid * 16 + r + 8 * lh;
      Out[(size_t)grow * kC + col] = acc[nt][r] + bv;
    }
  }
}

// =====================================================================
extern "C" void kernel_launch(void* const* d_in, const int* in_sizes, int n_in,
                              void* d_out, int out_size, void* d_ws, size_t ws_size,
                              hipStream_t stream) {
  (void)in_sizes; (void)n_in; (void)out_size; (void)ws_size;

  const float* query = (const float*)d_in[0];
  const float* key_  = (const float*)d_in[1];
  const float* value = (const float*)d_in[2];
  // d_in[3] = dense causal mask — causality is computed analytically instead.
  const float* Wq = (const float*)d_in[4];
  const float* bq = (const float*)d_in[5];
  const float* Wk = (const float*)d_in[6];
  const float* bk = (const float*)d_in[7];
  const float* Wv = (const float*)d_in[8];
  const float* bv = (const float*)d_in[9];
  const float* Wo = (const float*)d_in[10];
  const float* bo = (const float*)d_in[11];

  char* ws = (char*)d_ws;
  const size_t seg = (size_t)kB * kT * kC * sizeof(_Float16);  // 16 MB
  _Float16* qh = (_Float16*)(ws + 0 * seg);   // [B,NH,T,HS] f16
  _Float16* kh = (_Float16*)(ws + 1 * seg);
  _Float16* vh = (_Float16*)(ws + 2 * seg);
  _Float16* ah = (_Float16*)(ws + 3 * seg);   // [B,T,C] f16 attention output

  dim3 gGemm(kC / 64, (kB * kT) / 128);       // (16, 64)
  qkv_proj_kernel<<<gGemm, 256, 0, stream>>>(query, Wq, bq, qh);
  qkv_proj_kernel<<<gGemm, 256, 0, stream>>>(key_,  Wk, bk, kh);
  qkv_proj_kernel<<<gGemm, 256, 0, stream>>>(value, Wv, bv, vh);

  dim3 gAttn(kT / 64, kB * kNH);              // (32, 64)
  attn_kernel<<<gAttn, 128, 0, stream>>>(qh, kh, vh, ah);

  out_proj_kernel<<<gGemm, 256, 0, stream>>>(ah, Wo, bo, (float*)d_out);
}
